// MultiHeadAttentionLayer_89747636617314
// MI455X (gfx1250) — matrix operand order repaired and checked by reading, not compiled, not run on hardware
//
#include <hip/hip_runtime.h>

// ---------------------------------------------------------------------------
// MultiHeadAttention for MI455X (gfx1250, wave32, WMMA bf16 16x16x32)
//   B=4, S=2048, d_model=1024, H=16, dh=64
// Pipeline: qkv_proj (bf16 GEMM, V stored transposed)
//           -> flash attention (bf16 WMMA, f32 online softmax,
//              double-buffered GLOBAL_LOAD_ASYNC_TO_LDS_B128 K/V staging)
//           -> out_proj (bf16 GEMM, async A-tile staging, f32 output)
// ---------------------------------------------------------------------------

typedef __attribute__((ext_vector_type(16))) __bf16 v16bf;
typedef __attribute__((ext_vector_type(8)))  float  v8f;

#define DMODEL 1024
#define SEQ    2048
#define NBATCH 4
#define NHEAD  16
#define DHEAD  64

// LDS strides (elements). All rows 16-byte aligned so async B128 LDS writes
// are legal and fragment reads merge into ds_load_b128.
#define SA 40   // 128x32 A tiles (bf16): 80 B/row
#define SB 40   // transposed 32x128 B tiles (bf16): 80 B/row
#define SK 72   // K chunk [128 sk][64 d]: 144 B/row
#define SV 136  // V chunk [64 d][128 sk]: 272 B/row
#define SP 136  // P tile  [128 q][128 sk]: 272 B/row

// D = A*B + C, bf16 inputs, f32 accumulate (v_wmma_f32_16x16x32_bf16)
__device__ __forceinline__ v8f wmma_bf16(v16bf a, v16bf b, v8f c) {
  return __builtin_amdgcn_wmma_f32_16x16x32_bf16(
      /*neg_a=*/false, a, /*neg_b=*/false, b,
      /*c_mod=*/(short)0, c, /*reuse_a=*/false, /*reuse_b=*/false);
}

// Async global->LDS copy, 16 bytes per lane. VDST = LDS byte address (low 32
// bits of the flat pointer per the LDS aperture rule), VADDR = 64-bit global.
// Tracked by ASYNCcnt; completes in order with other async loads.
__device__ __forceinline__ void async_copy_b128(const void* gptr, void* lptr) {
  unsigned lds_off = (unsigned)(uintptr_t)lptr;
  asm volatile("global_load_async_to_lds_b128 %0, %1, off"
               :: "v"(lds_off), "v"(gptr) : "memory");
}
template <int N>
__device__ __forceinline__ void wait_async_le() {
  asm volatile("s_wait_asynccnt %0" :: "n"(N) : "memory");
}

// A fragment (16x32 bf16). base -> row 0 of the 16-row slice; K-contiguous rows.
// ISA: lane L holds row M=L%16; lanes 0-15: VGPR0-3 K=0..7, VGPR4-7 K=16..23;
// lanes 16-31: VGPR0-3 K=8..15, VGPR4-7 K=24..31.  Two 16B runs per lane.
__device__ __forceinline__ v16bf load_a_frag(const __bf16* base, int stride, int kbase) {
  int lane = threadIdx.x & 31;
  int hl   = lane >> 4;
  const __bf16* rp = base + (size_t)(lane & 15) * stride + kbase;
  union { v16bf v; unsigned u[8]; } f;
#pragma unroll
  for (int i = 0; i < 4; ++i)
    f.u[i] = *(const unsigned*)(rp + hl * 8 + 2 * i);
#pragma unroll
  for (int i = 0; i < 4; ++i)
    f.u[4 + i] = *(const unsigned*)(rp + 16 + hl * 8 + 2 * i);
  return f.v;
}

// B fragment (32x16 bf16) from K-contiguous LDS layout lds[n][k].
// Lane L: N=L%16; lanes 0-15 hold K=0..7,16..23, lanes 16-31 hold K=8..15,24..31.
// Two 16-byte runs per lane (16B-aligned -> 2x ds_load_b128).
__device__ __forceinline__ v16bf load_b_frag(const __bf16* base, int stride, int kbase) {
  int lane = threadIdx.x & 31;
  int hl   = lane >> 4;
  const __bf16* rp = base + (size_t)(lane & 15) * stride + kbase + hl * 8;
  union { v16bf v; unsigned u[8]; } f;
#pragma unroll
  for (int i = 0; i < 8; ++i)
    f.u[i] = *(const unsigned*)(rp + (i >> 2) * 16 + 2 * (i & 3));
  return f.v;
}

// ---------------------------------------------------------------------------
// Kernel 1: fused QKV projection.  grid = (8, 64, 3), block = 256 (8 waves).
// C[m][n] = sum_k X[m][k]*W[k][n] + bias[n].
//   Q, K written bf16 [B,H,S,dh];  V written bf16 TRANSPOSED [B,H,dh,S]
//   so the attention kernel can stage V with contiguous async B128 copies.
// ---------------------------------------------------------------------------
__global__ __launch_bounds__(256) void qkv_proj(
    const float* __restrict__ Xq, const float* __restrict__ Xk, const float* __restrict__ Xv,
    const float* __restrict__ Wq, const float* __restrict__ Wk, const float* __restrict__ Wv,
    const float* __restrict__ bq, const float* __restrict__ bk, const float* __restrict__ bv,
    unsigned short* __restrict__ Qo, unsigned short* __restrict__ Ko,
    unsigned short* __restrict__ Vo)
{
  __shared__ __bf16 As[128 * SA];
  __shared__ __bf16 Bs[128 * SB];

  const float* X; const float* W; const float* bias; __bf16* dst;
  int z = blockIdx.z;
  if (z == 0)      { X = Xq; W = Wq; bias = bq; dst = (__bf16*)Qo; }
  else if (z == 1) { X = Xk; W = Wk; bias = bk; dst = (__bf16*)Ko; }
  else             { X = Xv; W = Wv; bias = bv; dst = (__bf16*)Vo; }

  const int n0 = blockIdx.x * 128;
  const int m0 = blockIdx.y * 128;
  const int t = threadIdx.x, lane = t & 31, wid = t >> 5, hl = lane >> 4;
  const int waveM = wid >> 2, waveN = wid & 3;

  v8f vzero = {0.f, 0.f, 0.f, 0.f, 0.f, 0.f, 0.f, 0.f};
  v8f acc[4][2];
#pragma unroll
  for (int i = 0; i < 4; ++i)
#pragma unroll
    for (int j = 0; j < 2; ++j) acc[i][j] = vzero;

  for (int kt = 0; kt < DMODEL; kt += 32) {
    if (kt + 32 < DMODEL) {
      __builtin_prefetch(X + (size_t)(m0 + (t >> 1)) * DMODEL + kt + 32, 0, 1);
      __builtin_prefetch(W + (size_t)(kt + 32 + (t >> 4)) * DMODEL + n0, 0, 1);
    }
    // A tile 128x32 f32 -> bf16 LDS (needs conversion: VGPR path)
#pragma unroll
    for (int i = 0; i < 4; ++i) {
      int e = (i * 256 + t) * 4;
      int row = e >> 5, col = e & 31;
      float4 x4 = *(const float4*)(X + (size_t)(m0 + row) * DMODEL + kt + col);
      __bf16* p = &As[row * SA + col];
      p[0] = (__bf16)x4.x; p[1] = (__bf16)x4.y; p[2] = (__bf16)x4.z; p[3] = (__bf16)x4.w;
    }
    // B tile 32x128 f32 -> bf16 LDS transposed to [n][k]
#pragma unroll
    for (int i = 0; i < 4; ++i) {
      int e = (i * 256 + t) * 4;
      int kk = e >> 7, nn = e & 127;
      float4 w4 = *(const float4*)(W + (size_t)(kt + kk) * DMODEL + n0 + nn);
      Bs[(nn + 0) * SB + kk] = (__bf16)w4.x;
      Bs[(nn + 1) * SB + kk] = (__bf16)w4.y;
      Bs[(nn + 2) * SB + kk] = (__bf16)w4.z;
      Bs[(nn + 3) * SB + kk] = (__bf16)w4.w;
    }
    __syncthreads();

    v16bf af[4], bfr[2];
#pragma unroll
    for (int i = 0; i < 4; ++i)
      af[i] = load_a_frag(&As[(waveM * 64 + i * 16) * SA], SA, 0);
#pragma unroll
    for (int j = 0; j < 2; ++j)
      bfr[j] = load_b_frag(&Bs[(waveN * 32 + j * 16) * SB], SB, 0);
#pragma unroll
    for (int i = 0; i < 4; ++i)
#pragma unroll
      for (int j = 0; j < 2; ++j)
        acc[i][j] = wmma_bf16(af[i], bfr[j], acc[i][j]);
    __syncthreads();
  }

  // Epilogue: bias add, scatter bf16.
#pragma unroll
  for (int j = 0; j < 2; ++j) {
    int n = n0 + waveN * 32 + j * 16 + (lane & 15);
    float bvl = bias[n];
    int h = n >> 6, d = n & 63;
#pragma unroll
    for (int i = 0; i < 4; ++i) {
#pragma unroll
      for (int r = 0; r < 8; ++r) {
        int m = m0 + waveM * 64 + i * 16 + r + hl * 8;
        int b = m >> 11, s = m & (SEQ - 1);
        __bf16 val = (__bf16)(acc[i][j][r] + bvl);
        if (z == 2)  // V transposed: [B,H,dh,S]
          dst[(((size_t)(b * NHEAD + h)) * DHEAD + d) * SEQ + s] = val;
        else         // Q, K: [B,H,S,dh]
          dst[(((size_t)(b * NHEAD + h)) * SEQ + s) * DHEAD + d] = val;
      }
    }
  }
}

// ---------------------------------------------------------------------------
// Kernel 2: flash attention.  grid = (16 q-blocks, 64 bh), block = 256.
// Each wave owns 16 q-rows; streams Sk in chunks of 128 with online softmax.
// K/V chunks double-buffered with async B128 copies: chunk n+1's loads are
// issued before computing chunk n; s_wait_asynccnt 8 (in-order completion)
// guarantees chunk n's 8 per-wave copies have landed.
// Output written bf16 directly in concat layout [B, S, d_model].
// ---------------------------------------------------------------------------
__global__ __launch_bounds__(256) void attn_kernel(
    const unsigned short* __restrict__ Qb_, const unsigned short* __restrict__ Kb_,
    const unsigned short* __restrict__ Vt_, const float* __restrict__ mask,
    unsigned short* __restrict__ Ob_)
{
  const __bf16* Qb = (const __bf16*)Qb_;
  const __bf16* Kb = (const __bf16*)Kb_;   // [B,H,S,dh]
  const __bf16* Vt = (const __bf16*)Vt_;   // [B,H,dh,S]
  __bf16* Ob = (__bf16*)Ob_;

  __shared__ __bf16 Ks[2][128 * SK];  // [sk][d]  (d contiguous)
  __shared__ __bf16 Vs[2][64 * SV];   // [d][sk]  (sk contiguous)
  __shared__ __bf16 Ps[128 * SP];     // [q][sk]  (sk contiguous)

  const int bh = blockIdx.y;             // b*16 + h
  const int b = bh >> 4, h = bh & 15;
  const int q0 = blockIdx.x * 128;
  const int t = threadIdx.x, lane = t & 31, wid = t >> 5, hl = lane >> 4;
  const float scale = 0.125f;            // 1/sqrt(64)

  // Stage one 128-wide K/V chunk into LDS buffer `buf` (8 async ops/thread).
  auto stage = [&](int kc, int buf) {
#pragma unroll
    for (int i = 0; i < 4; ++i) {
      int e = (i * 256 + t) * 8;
      { // K: row = sk (0..127), col = d (multiple of 8)
        int row = e >> 6, col = e & 63;
        async_copy_b128(Kb + ((size_t)bh * SEQ + kc + row) * DHEAD + col,
                        &Ks[buf][row * SK + col]);
      }
      { // V: row = d (0..63), col = sk (multiple of 8)
        int row = e >> 7, col = e & 127;
        async_copy_b128(Vt + ((size_t)bh * DHEAD + row) * SEQ + kc + col,
                        &Vs[buf][row * SV + col]);
      }
    }
  };

  // Q fragments for this wave's 16 rows (reused across all k-chunks)
  const __bf16* qrow = Qb + ((size_t)bh * SEQ + q0 + wid * 16 + (lane & 15)) * DHEAD;
  v16bf aq[2];
#pragma unroll
  for (int f = 0; f < 2; ++f) {
    union { v16bf v; unsigned u[8]; } fa;
#pragma unroll
    for (int i = 0; i < 4; ++i)
      fa.u[i] = *(const unsigned*)(qrow + f * 32 + hl * 8 + 2 * i);
#pragma unroll
    for (int i = 0; i < 4; ++i)
      fa.u[4 + i] = *(const unsigned*)(qrow + f * 32 + 16 + hl * 8 + 2 * i);
    aq[f] = fa.v;
  }

  v8f vzero = {0.f, 0.f, 0.f, 0.f, 0.f, 0.f, 0.f, 0.f};
  float mrow[8], lrow[8];
  v8f accO[4];
#pragma unroll
  for (int r = 0; r < 8; ++r) { mrow[r] = -3.0e38f; lrow[r] = 0.f; }
#pragma unroll
  for (int dt = 0; dt < 4; ++dt) accO[dt] = vzero;

  stage(0, 0);
  for (int it = 0; it < SEQ / 128; ++it) {
    const int kc = it * 128;
    const int cur = it & 1;
    if (it + 1 < SEQ / 128) {
      stage(kc + 128, cur ^ 1);   // overlap next chunk's copy with this compute
      wait_async_le<8>();         // previous 8 ops (this chunk) have completed
    } else {
      wait_async_le<0>();
    }
    __syncthreads();

    const __bf16* Kc = Ks[cur];
    const __bf16* Vc = Vs[cur];

    // Scores: S = Q K^T (16 x 128 per wave), scaled, + mask * -1e9
    float sc[8][8];  // [n-tile][row]
#pragma unroll
    for (int nt = 0; nt < 8; ++nt) {
      v16bf bk0 = load_b_frag(&Kc[(nt * 16) * SK], SK, 0);
      v16bf bk1 = load_b_frag(&Kc[(nt * 16) * SK], SK, 32);
      v8f s = wmma_bf16(aq[0], bk0, vzero);
      s = wmma_bf16(aq[1], bk1, s);
      float mval = mask[b * SEQ + kc + nt * 16 + (lane & 15)] * -1.0e9f;
#pragma unroll
      for (int r = 0; r < 8; ++r) sc[nt][r] = s[r] * scale + mval;
    }

    // Online softmax: row r maps to q-row (wid*16 + r + 8*hl); the row's
    // columns live across the 16 lanes of this half-wave x 8 n-tiles.
    float mnew[8];
#pragma unroll
    for (int r = 0; r < 8; ++r) {
      mnew[r] = mrow[r];
#pragma unroll
      for (int nt = 0; nt < 8; ++nt) mnew[r] = fmaxf(mnew[r], sc[nt][r]);
    }
#pragma unroll
    for (int off = 1; off < 16; off <<= 1)
#pragma unroll
      for (int r = 0; r < 8; ++r)
        mnew[r] = fmaxf(mnew[r], __shfl_xor(mnew[r], off, 32));

    float fac[8], rsum[8];
#pragma unroll
    for (int r = 0; r < 8; ++r) {
      fac[r] = __expf(mrow[r] - mnew[r]);
      rsum[r] = 0.f;
      mrow[r] = mnew[r];
    }
#pragma unroll
    for (int nt = 0; nt < 8; ++nt)
#pragma unroll
      for (int r = 0; r < 8; ++r) {
        float p = __expf(sc[nt][r] - mnew[r]);
        rsum[r] += p;
        Ps[(wid * 16 + r + hl * 8) * SP + nt * 16 + (lane & 15)] = (__bf16)p;
      }
#pragma unroll
    for (int off = 1; off < 16; off <<= 1)
#pragma unroll
      for (int r = 0; r < 8; ++r) rsum[r] += __shfl_xor(rsum[r], off, 32);
#pragma unroll
    for (int r = 0; r < 8; ++r) lrow[r] = lrow[r] * fac[r] + rsum[r];
#pragma unroll
    for (int dt = 0; dt < 4; ++dt)
#pragma unroll
      for (int r = 0; r < 8; ++r) accO[dt][r] *= fac[r];
    __syncthreads();  // all P written before any wave consumes it

    // O += P @ V  (kdim = sk, 4 chunks of 32)
#pragma unroll
    for (int f = 0; f < 4; ++f) {
      v16bf pf = load_a_frag(&Ps[(wid * 16) * SP], SP, f * 32);
#pragma unroll
      for (int dt = 0; dt < 4; ++dt) {
        v16bf vf = load_b_frag(&Vc[(dt * 16) * SV], SV, f * 32);
        accO[dt] = wmma_bf16(pf, vf, accO[dt]);
      }
    }
    __syncthreads();  // all reads of Ks/Vs[cur]/Ps done before next iteration
  }

  // Normalize and write concat layout [B, S, d_model] bf16
#pragma unroll
  for (int dt = 0; dt < 4; ++dt) {
    int d = dt * 16 + (lane & 15);
#pragma unroll
    for (int r = 0; r < 8; ++r) {
      int q = q0 + wid * 16 + r + hl * 8;
      float o = accO[dt][r] / lrow[r];
      Ob[((size_t)b * SEQ + q) * DMODEL + h * DHEAD + d] = (__bf16)o;
    }
  }
}

// ---------------------------------------------------------------------------
// Kernel 3: output projection.  concat(bf16) @ Wo + bo -> f32 d_out.
// A tiles staged with async B128 copies; W tiles converted via VGPR path.
// grid = (8, 64), block = 256.
// ---------------------------------------------------------------------------
__global__ __launch_bounds__(256) void out_proj(
    const unsigned short* __restrict__ Ain_, const float* __restrict__ Wo,
    const float* __restrict__ bo, float* __restrict__ out)
{
  const __bf16* Ain = (const __bf16*)Ain_;
  __shared__ __bf16 As[128 * SA];
  __shared__ __bf16 Bs[128 * SB];

  const int n0 = blockIdx.x * 128;
  const int m0 = blockIdx.y * 128;
  const int t = threadIdx.x, lane = t & 31, wid = t >> 5, hl = lane >> 4;
  const int waveM = wid >> 2, waveN = wid & 3;

  v8f vzero = {0.f, 0.f, 0.f, 0.f, 0.f, 0.f, 0.f, 0.f};
  v8f acc[4][2];
#pragma unroll
  for (int i = 0; i < 4; ++i)
#pragma unroll
    for (int j = 0; j < 2; ++j) acc[i][j] = vzero;

  for (int kt = 0; kt < DMODEL; kt += 32) {
    if (kt + 32 < DMODEL)
      __builtin_prefetch(Wo + (size_t)(kt + 32 + (t >> 4)) * DMODEL + n0, 0, 1);
    // A tile 128x32 bf16: async global -> LDS, 2 x B128 per thread
#pragma unroll
    for (int i = 0; i < 2; ++i) {
      int e = (i * 256 + t) * 8;
      int row = e >> 5, col = e & 31;
      async_copy_b128(Ain + (size_t)(m0 + row) * DMODEL + kt + col,
                      &As[row * SA + col]);
    }
    // B tile 32x128 f32 -> bf16 LDS transposed to [n][k]
#pragma unroll
    for (int i = 0; i < 4; ++i) {
      int e = (i * 256 + t) * 4;
      int kk = e >> 7, nn = e & 127;
      float4 w4 = *(const float4*)(Wo + (size_t)(kt + kk) * DMODEL + n0 + nn);
      Bs[(nn + 0) * SB + kk] = (__bf16)w4.x;
      Bs[(nn + 1) * SB + kk] = (__bf16)w4.y;
      Bs[(nn + 2) * SB + kk] = (__bf16)w4.z;
      Bs[(nn + 3) * SB + kk] = (__bf16)w4.w;
    }
    wait_async_le<0>();
    __syncthreads();

    v16bf af[4], bfr[2];
#pragma unroll
    for (int i = 0; i < 4; ++i)
      af[i] = load_a_frag(&As[(waveM * 64 + i * 16) * SA], SA, 0);
#pragma unroll
    for (int j = 0; j < 2; ++j)
      bfr[j] = load_b_frag(&Bs[(waveN * 32 + j * 16) * SB], SB, 0);
#pragma unroll
    for (int i = 0; i < 4; ++i)
#pragma unroll
      for (int j = 0; j < 2; ++j)
        acc[i][j] = wmma_bf16(af[i], bfr[j], acc[i][j]);
    __syncthreads();
  }

#pragma unroll
  for (int j = 0; j < 2; ++j) {
    int n = n0 + waveN * 32 + j * 16 + (lane & 15);
    float bvl = bo[n];
#pragma unroll
    for (int i = 0; i < 4; ++i) {
#pragma unroll
      for (int r = 0; r < 8; ++r) {
        int m = m0 + waveM * 64 + i * 16 + r + hl * 8;
        out[(size_t)m * DMODEL + n] = acc[i][j][r] + bvl;
      }
    }
  }
}

// ---------------------------------------------------------------------------
// Launcher
// Inputs: 0:query 1:key 2:value 3:mask 4:Wq 5:bq 6:Wk 7:bk 8:Wv 9:bv 10:Wo 11:bo
// Workspace: Qb | Kb | Vt | attn-concat, each 8192*1024 bf16 (64 MB total).
// ---------------------------------------------------------------------------
extern "C" void kernel_launch(void* const* d_in, const int* in_sizes, int n_in,
                              void* d_out, int out_size, void* d_ws, size_t ws_size,
                              hipStream_t stream) {
  const float* query = (const float*)d_in[0];
  const float* key   = (const float*)d_in[1];
  const float* value = (const float*)d_in[2];
  const float* mask  = (const float*)d_in[3];
  const float* Wq    = (const float*)d_in[4];
  const float* bq    = (const float*)d_in[5];
  const float* Wk    = (const float*)d_in[6];
  const float* bk    = (const float*)d_in[7];
  const float* Wv    = (const float*)d_in[8];
  const float* bv    = (const float*)d_in[9];
  const float* Wo    = (const float*)d_in[10];
  const float* bo    = (const float*)d_in[11];

  const size_t NE = (size_t)NBATCH * SEQ * DMODEL;  // 8388608 elements
  unsigned short* Qb = (unsigned short*)d_ws;
  unsigned short* Kb = Qb + NE;
  unsigned short* Vt = Kb + NE;
  unsigned short* Ab = Vt + NE;

  dim3 blk(256);
  qkv_proj<<<dim3(DMODEL / 128, (NBATCH * SEQ) / 128, 3), blk, 0, stream>>>(
      query, key, value, Wq, Wk, Wv, bq, bk, bv, Qb, Kb, Vt);
  attn_kernel<<<dim3(SEQ / 128, NBATCH * NHEAD), blk, 0, stream>>>(
      Qb, Kb, Vt, mask, Ab);
  out_proj<<<dim3(DMODEL / 128, (NBATCH * SEQ) / 128), blk, 0, stream>>>(
      Ab, Wo, bo, (float*)d_out);
}
